// SKipPool_25890062861059
// MI455X (gfx1250) — matrix operand order repaired and checked by reading, not compiled, not run on hardware
//
#include <hip/hip_runtime.h>
#include <hip/hip_bf16.h>

typedef __attribute__((ext_vector_type(16))) __bf16 v16bf;
typedef __attribute__((ext_vector_type(8)))  __bf16 v8bf;
typedef __attribute__((ext_vector_type(8)))  float  v8f;

constexpr int   NN       = 3072;
constexpr int   DD       = 128;
constexpr int   KK       = 1536;            // top-k
constexpr int   FS       = 192;             // filter size
constexpr int   RES      = KK - FS + 1;     // 1345
constexpr int   ROWS_PAD = 1600;            // padded xk rows (max index used: 1551)
constexpr int   SORT_N   = 4096;
constexpr int   RT       = (RES + 15) / 16; // 85 row tiles
constexpr int   DT       = 13;              // diagonal tiles covering j in [0,192)
constexpr float TINYF    = 1.17549435e-38f;

// ---------------- kernel 1: scores = (x @ w + b) / ||w|| ----------------
__global__ void k_scores(const float* __restrict__ x, const float* __restrict__ w,
                         const float* __restrict__ b, float* __restrict__ scores) {
  int lane = threadIdx.x & 31;
  int wid  = threadIdx.x >> 5;
  int row  = blockIdx.x * 8 + wid;
  if (row >= NN) return;
  const float* xr = x + row * DD;
  float dot = 0.f, nrm = 0.f;
#pragma unroll
  for (int t = 0; t < DD / 32; ++t) {
    float wv = w[lane + t * 32];
    dot += xr[lane + t * 32] * wv;
    nrm += wv * wv;
  }
#pragma unroll
  for (int m = 16; m >= 1; m >>= 1) {
    dot += __shfl_xor(dot, m, 32);
    nrm += __shfl_xor(nrm, m, 32);
  }
  if (lane == 0) scores[row] = (dot + b[0]) * rsqrtf(nrm);
}

// ---------------- kernel 2: exact top-1536 (sorted desc, stable) ----------------
__global__ __launch_bounds__(1024) void k_topk(const float* __restrict__ scores,
                                               int* __restrict__ perm) {
  __shared__ unsigned long long key[SORT_N];
  int tid = threadIdx.x;
  for (int i = tid; i < SORT_N; i += 1024) {
    unsigned long long kk;
    if (i < NN) {
      unsigned u  = __float_as_uint(scores[i]);
      unsigned ou = (u & 0x80000000u) ? ~u : (u | 0x80000000u); // ascending orderable
      kk = ((unsigned long long)(~ou) << 32) | (unsigned)i;     // desc score, asc idx
    } else kk = ~0ull;
    key[i] = kk;
  }
  __syncthreads();
  for (int ksz = 2; ksz <= SORT_N; ksz <<= 1) {
    for (int jsz = ksz >> 1; jsz >= 1; jsz >>= 1) {
      for (int t = tid; t < SORT_N / 2; t += 1024) {
        int i = ((t & ~(jsz - 1)) << 1) | (t & (jsz - 1));
        int p = i | jsz;
        bool up = ((i & ksz) == 0);
        unsigned long long a = key[i], c = key[p];
        bool sw = up ? (a > c) : (a < c);
        if (sw) { key[i] = c; key[p] = a; }
      }
      __syncthreads();
    }
  }
  for (int i = tid; i < KK; i += 1024)
    perm[i] = (int)(unsigned)(key[i] & 0xFFFFFFFFull);
}

// ---------------- kernel 3: gather xk (f32 + bf16, zero-padded rows) ----------------
__global__ void k_gather(const float* __restrict__ x, const int* __restrict__ perm,
                         float* __restrict__ xk, unsigned short* __restrict__ xkbf) {
  int row = blockIdx.x;
  int d   = threadIdx.x;
  float v = 0.f;
  if (row < KK) v = x[perm[row] * DD + d];
  xk[row * DD + d] = v;
  unsigned u = __float_as_uint(v);
  unsigned r = (u + 0x7FFFu + ((u >> 16) & 1u)) >> 16; // RNE f32 -> bf16
  xkbf[row * DD + d] = (unsigned short)r;
}

// ---------------- kernel 4: a[i]=xk.wa, b[i]=xk.wb, sq[i]=||xk||^2 ----------------
__global__ void k_rowstats(const float* __restrict__ xk, const float* __restrict__ att,
                           float* __restrict__ av, float* __restrict__ bv,
                           float* __restrict__ sq) {
  int lane = threadIdx.x & 31;
  int wid  = threadIdx.x >> 5;
  int row  = blockIdx.x * 8 + wid;
  if (row >= ROWS_PAD) return;
  const float* xr = xk + row * DD;
  float da = 0.f, db = 0.f, ds = 0.f;
#pragma unroll
  for (int t = 0; t < DD / 32; ++t) {
    float xv = xr[lane + t * 32];
    da += xv * att[lane + t * 32];
    db += xv * att[DD + lane + t * 32];
    ds += xv * xv;
  }
#pragma unroll
  for (int m = 16; m >= 1; m >>= 1) {
    da += __shfl_xor(da, m, 32);
    db += __shfl_xor(db, m, 32);
    ds += __shfl_xor(ds, m, 32);
  }
  if (lane == 0) { av[row] = da; bv[row] = db; sq[row] = ds; }
}

// ---------------- kernel 5: banded Gram via WMMA + gk/sm epilogue ----------------
__global__ __launch_bounds__(32) void k_gram_wmma(
    const unsigned short* __restrict__ xkbf_us,
    const float* __restrict__ av, const float* __restrict__ bv,
    const float* __restrict__ sq,
    float* __restrict__ gk_buf, float* __restrict__ sm_buf) {
  const __bf16* xkbf = reinterpret_cast<const __bf16*>(xkbf_us);
  int tile = blockIdx.x;
  int r0   = (tile / DT) * 16;
  int dt   = tile % DT;
  int c0   = r0 + dt * 16;
  int lane = threadIdx.x;
  int lo   = lane & 15;
  int hi   = lane >> 4;

  const __bf16* rowA = xkbf + (r0 + lo) * DD; // A: M = lo (both half-waves)
  const __bf16* rowB = xkbf + (c0 + lo) * DD; // B: N = lo

  v8f acc = {0.f, 0.f, 0.f, 0.f, 0.f, 0.f, 0.f, 0.f};
#pragma unroll
  for (int k0 = 0; k0 < DD; k0 += 32) {
    // A 16x32 bf16 layout: lane holds K(k0+hi*16 .. +15) of row M=lo (contiguous)
    v16bf a  = *reinterpret_cast<const v16bf*>(rowA + k0 + hi * 16);
    // B 32x16 bf16 layout: lane holds K(k0+hi*16 .. +15) of column N=lo (contiguous)
    v16bf b = *reinterpret_cast<const v16bf*>(rowB + k0 + hi * 16);
    acc = __builtin_amdgcn_wmma_f32_16x16x32_bf16(false, a, false, b,
                                                  (short)0, acc, false, false);
  }

  int   c   = c0 + lo;
  float sqc = sq[c];
  float bc  = bv[c];
#pragma unroll
  for (int v = 0; v < 8; ++v) {
    int r = r0 + v + hi * 8;  // C layout: M = v + 8*hi, N = lane&15
    int j = c - r;
    if (r < RES && j >= 0 && j < FS) {
      float g2   = acc[v];
      float s2   = sq[r] + sqc - 2.0f * g2;
      float dist = sqrtf(fmaxf(s2, 1e-24f));
      float gk   = expf(-dist * 0.5f) - 1e-5f; // 2*sigma^2 = 2
      gk_buf[r * FS + j] = gk;
      sm_buf[r * FS + j] = av[r] + bc;
    }
  }
}

// ---------------- kernel 6: per-row softmax(sm) + stable sort by gk desc ----------------
__global__ __launch_bounds__(256) void k_softmax_sort(
    const float* __restrict__ sm_buf, const float* __restrict__ gk_buf,
    float* __restrict__ sm_sort, float* __restrict__ gk_sort) {
  __shared__ float red[256];
  __shared__ float s_sm[256];
  __shared__ float s_gk[256];
  __shared__ unsigned long long s_key[256];
  int row = blockIdx.x;
  int tid = threadIdx.x;

  float smv = (tid < FS) ? sm_buf[row * FS + tid] : -INFINITY;
  red[tid] = smv; __syncthreads();
  for (int s = 128; s > 0; s >>= 1) { if (tid < s) red[tid] = fmaxf(red[tid], red[tid + s]); __syncthreads(); }
  float mx = red[0]; __syncthreads();
  float e = (tid < FS) ? expf(smv - mx) : 0.f;
  red[tid] = e; __syncthreads();
  for (int s = 128; s > 0; s >>= 1) { if (tid < s) red[tid] += red[tid + s]; __syncthreads(); }
  float inv = 1.0f / red[0];
  s_sm[tid] = e * inv;

  float g = (tid < FS) ? gk_buf[row * FS + tid] : 0.f;
  s_gk[tid] = g;
  unsigned long long kk;
  if (tid < FS) {
    unsigned u  = __float_as_uint(g);
    unsigned ou = (u & 0x80000000u) ? ~u : (u | 0x80000000u);
    kk = ((unsigned long long)(~ou) << 32) | (unsigned)tid; // desc gk, stable ties
  } else kk = ~0ull;
  s_key[tid] = kk;
  __syncthreads();

  for (int ksz = 2; ksz <= 256; ksz <<= 1) {
    for (int jsz = ksz >> 1; jsz >= 1; jsz >>= 1) {
      if (tid < 128) {
        int i = ((tid & ~(jsz - 1)) << 1) | (tid & (jsz - 1));
        int p = i | jsz;
        bool up = ((i & ksz) == 0);
        unsigned long long a = s_key[i], c = s_key[p];
        bool sw = up ? (a > c) : (a < c);
        if (sw) { s_key[i] = c; s_key[p] = a; }
      }
      __syncthreads();
    }
  }
  if (tid < FS) {
    int idx = (int)(unsigned)(s_key[tid] & 0xFFFFFFFFull);
    gk_sort[row * FS + tid] = s_gk[idx];
    sm_sort[row * FS + tid] = s_sm[idx];
  }
}

// ---------------- kernel 7: logits_col[j] = sum_r sm_sort[r][j] ----------------
__global__ void k_colsum(const float* __restrict__ sm_sort, float* __restrict__ logits_col) {
  int col = threadIdx.x;
  float s = 0.f;
  for (int r = 0; r < RES; ++r) s += sm_sort[r * FS + col];
  logits_col[col] = s;
}

// ---------------- kernel 8: threefry gumbel + gumbel-softmax + multiply ----------------
__device__ __forceinline__ unsigned rotl32(unsigned x, int r) {
  return (x << r) | (x >> (32 - r));
}
__device__ __forceinline__ void threefry2x32(unsigned ks0, unsigned ks1,
                                             unsigned x0, unsigned x1,
                                             unsigned& o0, unsigned& o1) {
  unsigned ks2 = ks0 ^ ks1 ^ 0x1BD11BDAu;
  x0 += ks0; x1 += ks1;
  const int r0[4] = {13, 15, 26, 6};
  const int r1[4] = {17, 29, 16, 24};
#pragma unroll
  for (int i = 0; i < 4; ++i) { x0 += x1; x1 = rotl32(x1, r0[i]); x1 ^= x0; }
  x0 += ks1; x1 += ks2 + 1u;
#pragma unroll
  for (int i = 0; i < 4; ++i) { x0 += x1; x1 = rotl32(x1, r1[i]); x1 ^= x0; }
  x0 += ks2; x1 += ks0 + 2u;
#pragma unroll
  for (int i = 0; i < 4; ++i) { x0 += x1; x1 = rotl32(x1, r0[i]); x1 ^= x0; }
  x0 += ks0; x1 += ks1 + 3u;
#pragma unroll
  for (int i = 0; i < 4; ++i) { x0 += x1; x1 = rotl32(x1, r1[i]); x1 ^= x0; }
  x0 += ks1; x1 += ks2 + 4u;
#pragma unroll
  for (int i = 0; i < 4; ++i) { x0 += x1; x1 = rotl32(x1, r0[i]); x1 ^= x0; }
  x0 += ks2; x1 += ks0 + 5u;
  o0 = x0; o1 = x1;
}

__global__ __launch_bounds__(256) void k_final(
    const float* __restrict__ gk_sort, const float* __restrict__ logits_col,
    const int* __restrict__ epoch_p, float* __restrict__ out) {
  __shared__ float red[256];
  int row = blockIdx.x;
  int tid = threadIdx.x;
  float tau0 = 10.0f * powf(0.01f, (float)epoch_p[0] * 0.01f);
  float z = -INFINITY;
  if (tid < FS) {
    int m = row * FS + tid;
    const int half = (RES * FS) / 2;
    unsigned x0, x1, o0, o1;
    if (m < half) { x0 = (unsigned)m; x1 = (unsigned)(m + half); }
    else          { x0 = (unsigned)(m - half); x1 = (unsigned)m; }
    threefry2x32(0u, 777u, x0, x1, o0, o1);
    unsigned bits = (m < half) ? o0 : o1;
    float f = __uint_as_float((bits >> 9) | 0x3F800000u) - 1.0f; // [0,1)
    float u = fmaxf(TINYF, f * (1.0f - TINYF) + TINYF);
    float g = -logf(-logf(u));
    float logit = (tid == 0) ? -INFINITY : logits_col[tid];
    z = (g + logit) / tau0;
  }
  red[tid] = z; __syncthreads();
  for (int s = 128; s > 0; s >>= 1) { if (tid < s) red[tid] = fmaxf(red[tid], red[tid + s]); __syncthreads(); }
  float mx = red[0]; __syncthreads();
  float e = (tid < FS) ? expf(z - mx) : 0.f;
  red[tid] = e; __syncthreads();
  for (int s = 128; s > 0; s >>= 1) { if (tid < s) red[tid] += red[tid + s]; __syncthreads(); }
  float inv = 1.0f / red[0];
  if (tid < FS) out[row * FS + tid] = gk_sort[row * FS + tid] * (e * inv);
}

// ---------------- launch ----------------
extern "C" void kernel_launch(void* const* d_in, const int* in_sizes, int n_in,
                              void* d_out, int out_size, void* d_ws, size_t ws_size,
                              hipStream_t stream) {
  const float* x       = (const float*)d_in[0];
  // d_in[1] = edge_index (unused by the reference)
  const float* score_w = (const float*)d_in[2];
  const float* score_b = (const float*)d_in[3];
  const float* att     = (const float*)d_in[4];
  const int*   epoch   = (const int*)d_in[5];
  float*       out     = (float*)d_out;

  char* ws = (char*)d_ws;
  float*          scores     = (float*)(ws + 0);          //  3072 f32
  int*            perm       = (int*)(ws + 12288);        //  1536 i32
  float*          xk         = (float*)(ws + 18432);      //  1600*128 f32
  unsigned short* xkbf       = (unsigned short*)(ws + 837632);   // 1600*128 bf16
  float*          av         = (float*)(ws + 1247232);    //  1600 f32
  float*          bv         = (float*)(ws + 1253632);    //  1600 f32
  float*          sq         = (float*)(ws + 1260032);    //  1600 f32
  float*          gk_buf     = (float*)(ws + 1266432);    //  1345*192 f32
  float*          sm_buf     = (float*)(ws + 2299392);    //  1345*192 f32
  float*          gk_sort    = (float*)(ws + 3332352);    //  1345*192 f32
  float*          sm_sort    = (float*)(ws + 4365312);    //  1345*192 f32
  float*          logits_col = (float*)(ws + 5398272);    //  192 f32

  k_scores      <<<NN / 8,        256, 0, stream>>>(x, score_w, score_b, scores);
  k_topk        <<<1,            1024, 0, stream>>>(scores, perm);
  k_gather      <<<ROWS_PAD,      DD,  0, stream>>>(x, perm, xk, xkbf);
  k_rowstats    <<<ROWS_PAD / 8, 256,  0, stream>>>(xk, att, av, bv, sq);
  k_gram_wmma   <<<RT * DT,       32,  0, stream>>>(xkbf, av, bv, sq, gk_buf, sm_buf);
  k_softmax_sort<<<RES,          256,  0, stream>>>(sm_buf, gk_buf, sm_sort, gk_sort);
  k_colsum      <<<1,             FS,  0, stream>>>(sm_sort, logits_col);
  k_final       <<<RES,          256,  0, stream>>>(gk_sort, logits_col, epoch, out);
}
